// VQVAE_22600117912203
// MI455X (gfx1250) — compile-verified, matrix operand order checked by reading, not run on hardware
//
#include <hip/hip_runtime.h>

// ---------- types ----------
typedef __bf16 bf16_t;
typedef __attribute__((ext_vector_type(16))) __bf16 v16bf;
typedef __attribute__((ext_vector_type(8)))  __bf16 bf16x8;
typedef __attribute__((ext_vector_type(8)))  float  v8f;

// =====================================================================
// Implicit-GEMM WMMA kernel (bf16 x bf16 -> f32 accumulate)
//   mode 0: A is plain [M,K] row-major (stride lda)
//   mode 1: A is NHWC activation, im2col on the fly (conv, stride 1, pad)
//   mode 2: A is NHWC activation, convT gather (k=ksz, stride 2, pad)
// B weights are pre-transposed: BwT[Npad][K] row-major.
// Block: 256 threads (8 waves), tile 64x64, K step 64 (2 slabs of 32).
// Each wave owns two 16x16 output tiles -> 4 WMMAs per barrier pair.
// Loop structure: outer over conv taps (addressing hoisted), inner over
// 64-wide channel slabs (pure pointer increments).
// reluFlag: 0=none, 1=relu both outputs, 2=relu only bf16 output
// =====================================================================
__global__ __launch_bounds__(256) void k_conv_gemm(
    const bf16_t* __restrict__ A, const bf16_t* __restrict__ BwT,
    const float* __restrict__ bias, const float* __restrict__ resid, int ldr,
    float* __restrict__ outF, int ldf, bf16_t* __restrict__ outH, int ldh,
    int M, int Npad, int Nreal, int K,
    int mode, int lda,
    int Cin, int IH, int IW, int OH, int OW, int ksz, int pad, int reluFlag)
{
    __shared__ __align__(16) bf16_t As[64][72];
    __shared__ __align__(16) bf16_t Bs[64][72];

    const int tid  = threadIdx.x;
    const int lane = tid & 31;
    const int wave = tid >> 5;
    const int wr   = wave & 3;       // 4 row tiles of 16
    const int wc   = wave >> 2;      // 2 col groups of 32
    const int row0 = blockIdx.y * 64;
    const int col0 = blockIdx.x * 64;

    v8f acc0 = {0.f,0.f,0.f,0.f,0.f,0.f,0.f,0.f};
    v8f acc1 = {0.f,0.f,0.f,0.f,0.f,0.f,0.f,0.f};

    // tile loader assignment: 64 rows x 64 k, 16 bf16 per thread
    const int ldRow = tid >> 2;          // 0..63
    const int ldK   = (tid & 3) * 16;    // 0,16,32,48
    const int arow  = row0 + ldRow;

    int b = 0, oy = 0, ox = 0;
    if (mode != 0) {
        const int hw = OH * OW;
        b = arow / hw;
        const int rr = arow - b * hw;
        oy = rr / OW; ox = rr - oy * OW;
    }

    // fragment coordinates (fixed per wave/lane)
    const int khalf = (lane >> 4) << 3;          // 0 or 8
    const int ma    = (wr << 4) + (lane & 15);
    const int nb0   = (wc << 5) + (lane & 15);

    const int ntaps  = (mode == 0) ? 1 : (K / Cin);
    const int CinEff = (mode == 0) ? K : Cin;
    const int cin64  = CinEff >> 6;

    for (int tap = 0; tap < ntaps; ++tap) {
        // ---- per-tap address setup (all div/branch logic lives here) ----
        bool valid = true;
        const bf16_t* aBase;
        if (mode == 0) {
            aBase = A + (size_t)arow * lda + ldK;
        } else {
            const int ky = tap / ksz, kx = tap - ky * ksz;
            int iy, ix;
            if (mode == 1) {
                iy = oy + ky - pad; ix = ox + kx - pad;
                valid = (iy >= 0) & (iy < IH) & (ix >= 0) & (ix < IW);
            } else {
                const int ty = oy + pad - ky, tx = ox + pad - kx;
                valid = (ty >= 0) & ((ty & 1) == 0) & (tx >= 0) & ((tx & 1) == 0);
                iy = ty >> 1; ix = tx >> 1;
                valid = valid & (iy < IH) & (ix < IW);
            }
            aBase = A + (((size_t)b * IH + iy) * IW + ix) * (size_t)Cin + ldK;
        }
        const bf16_t* bBase = BwT + (size_t)(col0 + ldRow) * K
                                  + (size_t)tap * CinEff + ldK;

        // ---- hot loop: 64-wide channel slabs ----
        for (int cs = 0; cs < cin64; ++cs) {
            const bf16_t* ap = aBase + (cs << 6);
            const bf16_t* bp = bBase + (cs << 6);
            bf16x8 a0, a1;
            if (valid) {
                a0 = *reinterpret_cast<const bf16x8*>(ap);
                a1 = *reinterpret_cast<const bf16x8*>(ap + 8);
                __builtin_prefetch(ap + 64, 0, 0);
            } else {
                #pragma unroll
                for (int i = 0; i < 8; ++i) { a0[i] = (bf16_t)0.f; a1[i] = (bf16_t)0.f; }
            }
            bf16x8 b0v = *reinterpret_cast<const bf16x8*>(bp);
            bf16x8 b1v = *reinterpret_cast<const bf16x8*>(bp + 8);
            __builtin_prefetch(bp + 64, 0, 0);

            *reinterpret_cast<bf16x8*>(&As[ldRow][ldK])     = a0;
            *reinterpret_cast<bf16x8*>(&As[ldRow][ldK + 8]) = a1;
            *reinterpret_cast<bf16x8*>(&Bs[ldRow][ldK])     = b0v;
            *reinterpret_cast<bf16x8*>(&Bs[ldRow][ldK + 8]) = b1v;
            __syncthreads();

            #pragma unroll
            for (int s = 0; s < 2; ++s) {
                const int kh = (s << 5) + khalf;
                bf16x8 alo  = *reinterpret_cast<const bf16x8*>(&As[ma][kh]);
                bf16x8 ahi  = *reinterpret_cast<const bf16x8*>(&As[ma][kh + 16]);
                bf16x8 blo0 = *reinterpret_cast<const bf16x8*>(&Bs[nb0][kh]);
                bf16x8 bhi0 = *reinterpret_cast<const bf16x8*>(&Bs[nb0][kh + 16]);
                bf16x8 blo1 = *reinterpret_cast<const bf16x8*>(&Bs[nb0 + 16][kh]);
                bf16x8 bhi1 = *reinterpret_cast<const bf16x8*>(&Bs[nb0 + 16][kh + 16]);
                v16bf af, bf0, bf1;
                #pragma unroll
                for (int i = 0; i < 8; ++i) {
                    af[i]  = alo[i];   af[i + 8]  = ahi[i];
                    bf0[i] = blo0[i];  bf0[i + 8] = bhi0[i];
                    bf1[i] = blo1[i];  bf1[i + 8] = bhi1[i];
                }
                acc0 = __builtin_amdgcn_wmma_f32_16x16x32_bf16(
                    false, af, false, bf0, (short)0, acc0, false, false);
                acc1 = __builtin_amdgcn_wmma_f32_16x16x32_bf16(
                    false, af, false, bf1, (short)0, acc1, false, false);
            }
            __syncthreads();
        }
    }

    // ---- epilogue: bias / residual / relu / dual-precision store ----
    const int n0    = col0 + (wc << 5) + (lane & 15);
    const int n1    = n0 + 16;
    const int mbase = row0 + (wr << 4) + ((lane >> 4) << 3);
    const float bs0 = (bias && n0 < Nreal) ? bias[n0] : 0.f;
    const float bs1 = (bias && n1 < Nreal) ? bias[n1] : 0.f;
    #pragma unroll
    for (int v = 0; v < 8; ++v) {
        const int m = mbase + v;
        if (n0 < Nreal) {
            float x = acc0[v] + bs0;
            if (resid) x += resid[(size_t)m * ldr + n0];
            float xf = x, xh = x;
            if (reluFlag == 1) { xf = fmaxf(x, 0.f); xh = xf; }
            else if (reluFlag == 2) { xh = fmaxf(x, 0.f); }
            if (outF) outF[(size_t)m * ldf + n0] = xf;
            if (outH) outH[(size_t)m * ldh + n0] = (bf16_t)xh;
        }
        if (n1 < Nreal) {
            float x = acc1[v] + bs1;
            if (resid) x += resid[(size_t)m * ldr + n1];
            float xf = x, xh = x;
            if (reluFlag == 1) { xf = fmaxf(x, 0.f); xh = xf; }
            else if (reluFlag == 2) { xh = fmaxf(x, 0.f); }
            if (outF) outF[(size_t)m * ldf + n1] = xf;
            if (outH) outH[(size_t)m * ldh + n1] = (bf16_t)xh;
        }
    }
}

// ---------- elementwise / transform kernels ----------
__global__ void k_cast_bf16(const float* __restrict__ s, bf16_t* __restrict__ d, size_t n) {
    size_t i = (size_t)blockIdx.x * blockDim.x + threadIdx.x;
    if (i < n) d[i] = (bf16_t)s[i];
}

// conv weights [O,I,KH,KW] -> wtT[o][(ky*KW+kx)*I + i]   (layout [Npad][K])
__global__ void k_wt_conv(const float* __restrict__ w, bf16_t* __restrict__ wt,
                          int O, int I, int KH, int KW, int Npad) {
    const size_t K = (size_t)KH * KW * I;
    size_t idx = (size_t)blockIdx.x * blockDim.x + threadIdx.x;
    if (idx >= (size_t)Npad * K) return;
    int o = (int)(idx / K);
    size_t kidx = idx - (size_t)o * K;
    int tap = (int)(kidx / I);
    int i = (int)(kidx - (size_t)tap * I);
    int ky = tap / KW, kx = tap - ky * KW;
    float v = (o < O) ? w[(((size_t)o * I + i) * KH + ky) * KW + kx] : 0.f;
    wt[idx] = (bf16_t)v;
}

// convT weights [I,O,KH,KW] -> wtT[o][(ky*KW+kx)*I + i]  (layout [Npad][K])
__global__ void k_wt_convT(const float* __restrict__ w, bf16_t* __restrict__ wt,
                           int I, int O, int KH, int KW, int Npad) {
    const size_t K = (size_t)KH * KW * I;
    size_t idx = (size_t)blockIdx.x * blockDim.x + threadIdx.x;
    if (idx >= (size_t)Npad * K) return;
    int o = (int)(idx / K);
    size_t kidx = idx - (size_t)o * K;
    int tap = (int)(kidx / I);
    int i = (int)(kidx - (size_t)tap * I);
    int ky = tap / KW, kx = tap - ky * KW;
    float v = (o < O) ? w[(((size_t)i * O + o) * KH + ky) * KW + kx] : 0.f;
    wt[idx] = (bf16_t)v;
}

__global__ void k_cnorm(const float* __restrict__ cb, int Kc, int C, float* __restrict__ out) {
    int k = blockIdx.x * blockDim.x + threadIdx.x;
    if (k >= Kc) return;
    float s = 0.f;
    for (int c = 0; c < C; ++c) { float v = cb[(size_t)k * C + c]; s += v * v; }
    out[k] = s;
}

// fusedF[r*ld + c] = relu(fusedF[r*ld + c] + txt[(r>>8)*C + c])
__global__ void k_txt_relu(float* __restrict__ fusedF, const float* __restrict__ txt,
                           int C, int ld, size_t total) {
    size_t i = (size_t)blockIdx.x * blockDim.x + threadIdx.x;
    if (i >= total) return;
    size_t r = i / C; int c = (int)(i - r * C);
    int b = (int)(r >> 8);   // 256 pixels per batch
    float v = fusedF[r * ld + c] + txt[(size_t)b * C + c];
    fusedF[r * ld + c] = fmaxf(v, 0.f);
}

// argmin over codebook distances; d_k = |c_k|^2 - 2 z.c_k (|z|^2 irrelevant)
__global__ void k_argmin(const float* __restrict__ scores, const float* __restrict__ cnorm,
                         int M, int Kc, int* __restrict__ ids) {
    int row = blockIdx.x * (blockDim.x >> 5) + (threadIdx.x >> 5);
    if (row >= M) return;
    int lane = threadIdx.x & 31;
    float best = 3.4e38f; int bi = 0x7fffffff;
    const float* sr = scores + (size_t)row * Kc;
    for (int k = lane; k < Kc; k += 32) {
        float d = cnorm[k] - 2.f * sr[k];
        if (d < best) { best = d; bi = k; }
    }
    for (int off = 16; off; off >>= 1) {
        float ob = __shfl_xor(best, off, 32);
        int   oi = __shfl_xor(bi,   off, 32);
        if (ob < best || (ob == best && oi < bi)) { best = ob; bi = oi; }
    }
    if (lane == 0) ids[row] = bi;
}

__global__ void k_count(const int* __restrict__ ids, float* __restrict__ counts, int M) {
    int i = blockIdx.x * blockDim.x + threadIdx.x;
    if (i < M) atomicAdd(&counts[ids[i]], 1.0f);
}

// gather zq = codebook[ids], accumulate sum (zq-z)^2, optional f32/bf16 outputs
__global__ __launch_bounds__(256) void k_vq_gather(
    const int* __restrict__ ids, const float* __restrict__ cb, int C,
    const float* __restrict__ zF, int ldz,
    float* __restrict__ qF, int ldqf, bf16_t* __restrict__ qH, int ldqh,
    float* __restrict__ sqAccum)
{
    int r = blockIdx.x;
    int id = ids[r];
    const float* cv = cb + (size_t)id * C;
    float local = 0.f;
    for (int c = threadIdx.x; c < C; c += blockDim.x) {
        float q = cv[c];
        float z = zF[(size_t)r * ldz + c];
        float d = q - z; local += d * d;
        if (qF) qF[(size_t)r * ldqf + c] = q;
        if (qH) qH[(size_t)r * ldqh + c] = (bf16_t)q;
    }
    __shared__ float red[256];
    red[threadIdx.x] = local; __syncthreads();
    for (int s = 128; s; s >>= 1) {
        if (threadIdx.x < (unsigned)s) red[threadIdx.x] += red[threadIdx.x + s];
        __syncthreads();
    }
    if (threadIdx.x == 0) atomicAdd(sqAccum, red[0]);
}

// acc layout: [0]=sq_top [1]=sq_bot [2..513]=counts_top [514..1025]=counts_bot
__global__ __launch_bounds__(512) void k_finalize(const float* __restrict__ acc,
                                                  float* __restrict__ out) {
    __shared__ float st[512], sb[512];
    int t = threadIdx.x;
    float ct  = acc[2 + t]   / 8192.f;
    float cbv = acc[514 + t] / 32768.f;
    st[t] = ct  * __logf(ct  + 1e-10f);
    sb[t] = cbv * __logf(cbv + 1e-10f);
    __syncthreads();
    for (int s = 256; s; s >>= 1) {
        if (t < s) { st[t] += st[t + s]; sb[t] += sb[t + s]; }
        __syncthreads();
    }
    if (t == 0) {
        float loss = 1.25f * (acc[0] / (8192.f * 256.f) + acc[1] / (32768.f * 512.f));
        out[0] = loss;
        out[1] = __expf(-st[0]);
        out[2] = __expf(-sb[0]);
    }
}

// NHWC [B*56*56, 3] -> NCHW [B,3,56,56]
__global__ void k_nchw(const float* __restrict__ src, float* __restrict__ dst, int total) {
    int i = blockIdx.x * blockDim.x + threadIdx.x;
    if (i >= total) return;
    int r = i / 3, c = i - r * 3;
    int b = r / 3136, p = r - b * 3136;
    dst[((size_t)b * 3 + c) * 3136 + p] = src[i];
}

// =====================================================================
extern "C" void kernel_launch(void* const* d_in, const int* in_sizes, int n_in,
                              void* d_out, int out_size, void* d_ws, size_t ws_size,
                              hipStream_t stream) {
    (void)in_sizes; (void)n_in; (void)out_size; (void)ws_size;
    // ---- inputs ----
    const float* lve     = (const float*)d_in[0];   // [128,256,1024]  (NHWC-ready)
    const float* text    = (const float*)d_in[1];   // [128,768]
    const float* qry     = (const float*)d_in[2];   // [128,64,256]    (NHWC-ready)
    const float* cb_top  = (const float*)d_in[3];   // [512,256]
    const float* cb_bot  = (const float*)d_in[4];   // [512,512]
    const float* tb_w    = (const float*)d_in[5];  const float* tb_b    = (const float*)d_in[6];
    const float* bvis_w  = (const float*)d_in[7];  const float* bvis_b  = (const float*)d_in[8];
    const float* btxt_w  = (const float*)d_in[9];  const float* btxt_b  = (const float*)d_in[10];
    const float* dtin_w  = (const float*)d_in[11]; const float* dtin_b  = (const float*)d_in[12];
    const float* dtr1_w  = (const float*)d_in[13]; const float* dtr1_b  = (const float*)d_in[14];
    const float* dtr2_w  = (const float*)d_in[15]; const float* dtr2_b  = (const float*)d_in[16];
    const float* dtup_w  = (const float*)d_in[17]; const float* dtup_b  = (const float*)d_in[18];
    const float* up_w    = (const float*)d_in[19]; const float* up_b    = (const float*)d_in[20];
    const float* ds_w    = (const float*)d_in[21]; const float* ds_b    = (const float*)d_in[22];
    const float* dbin_w  = (const float*)d_in[23]; const float* dbin_b  = (const float*)d_in[24];
    const float* dbr1_w  = (const float*)d_in[25]; const float* dbr1_b  = (const float*)d_in[26];
    const float* dbr2_w  = (const float*)d_in[27]; const float* dbr2_b  = (const float*)d_in[28];
    const float* dbu1_w  = (const float*)d_in[29]; const float* dbu1_b  = (const float*)d_in[30];
    const float* dbu2_w  = (const float*)d_in[31]; const float* dbu2_b  = (const float*)d_in[32];
    float* out = (float*)d_out;

    // ---- workspace bump allocator ----
    char* W = (char*)d_ws;
    size_t off = 0;
    auto WA = [&](size_t bytes) -> void* {
        void* p = W + off;
        off = (off + bytes + 255) & ~(size_t)255;
        return p;
    };
    bf16_t* lve_bf  = (bf16_t*)WA((size_t)33554432 * 2);
    bf16_t* txt_bf  = (bf16_t*)WA((size_t)98304 * 2);
    bf16_t* q2d_bf  = (bf16_t*)WA((size_t)2097152 * 2);
    bf16_t* wt_txt  = (bf16_t*)WA((size_t)256 * 768 * 2);
    bf16_t* wt_vis  = (bf16_t*)WA((size_t)256 * 1024 * 2);
    bf16_t* wt_tb   = (bf16_t*)WA((size_t)256 * 2304 * 2);
    bf16_t* wt_dtin = (bf16_t*)WA((size_t)512 * 2304 * 2);
    bf16_t* wt_dtr1 = (bf16_t*)WA((size_t)512 * 4608 * 2);
    bf16_t* wt_dtr2 = (bf16_t*)WA((size_t)512 * 4608 * 2);
    bf16_t* wt_dtup = (bf16_t*)WA((size_t)256 * 8192 * 2);
    bf16_t* wt_up   = (bf16_t*)WA((size_t)256 * 4096 * 2);
    bf16_t* wt_ds   = (bf16_t*)WA((size_t)768 * 6912 * 2);
    bf16_t* wt_dbin = (bf16_t*)WA((size_t)768 * 6912 * 2);
    bf16_t* wt_dbr1 = (bf16_t*)WA((size_t)768 * 6912 * 2);
    bf16_t* wt_dbr2 = (bf16_t*)WA((size_t)768 * 6912 * 2);
    bf16_t* wt_dbu1 = (bf16_t*)WA((size_t)384 * 12288 * 2);
    bf16_t* wt_dbu2 = (bf16_t*)WA((size_t)64 * 6144 * 2);
    bf16_t* cbH_top = (bf16_t*)WA((size_t)512 * 256 * 2);
    bf16_t* cbH_bot = (bf16_t*)WA((size_t)512 * 512 * 2);
    float* cn_top = (float*)WA(512 * 4);
    float* cn_bot = (float*)WA(512 * 4);
    float* accb   = (float*)WA(1026 * 4);
    float* txtF   = (float*)WA((size_t)128 * 256 * 4);
    float*  zetopF = (float*)WA((size_t)2097152 * 4);
    bf16_t* zetopH = (bf16_t*)WA((size_t)2097152 * 2);
    float*  scores = (float*)WA((size_t)32768 * 512 * 4);
    int*    ids    = (int*)WA((size_t)32768 * 4);
    float*  zqtopF = (float*)WA((size_t)2097152 * 4);
    bf16_t* zqtopH = (bf16_t*)WA((size_t)2097152 * 2);
    float*  hTop   = (float*)WA((size_t)8192 * 512 * 4);
    bf16_t* aTop   = (bf16_t*)WA((size_t)8192 * 512 * 2);
    bf16_t* bTop   = (bf16_t*)WA((size_t)8192 * 512 * 2);
    bf16_t* cTop   = (bf16_t*)WA((size_t)8192 * 512 * 2);
    float*  fusedF = (float*)WA((size_t)32768 * 512 * 4);
    bf16_t* fusedH = (bf16_t*)WA((size_t)32768 * 512 * 2);
    bf16_t* zqH    = (bf16_t*)WA((size_t)32768 * 768 * 2);
    bf16_t* zq2H   = (bf16_t*)WA((size_t)25088 * 768 * 2);
    float*  hBot   = (float*)WA((size_t)25088 * 768 * 4);
    bf16_t* aBot   = (bf16_t*)WA((size_t)25088 * 768 * 2);
    bf16_t* bBot   = (bf16_t*)WA((size_t)25088 * 768 * 2);
    bf16_t* cBot   = (bf16_t*)WA((size_t)25088 * 768 * 2);
    bf16_t* up1H   = (bf16_t*)WA((size_t)100352 * 384 * 2);
    float*  reconF = (float*)WA((size_t)401408 * 3 * 4);

    auto blocks = [](size_t n, int bs) { return dim3((unsigned)((n + bs - 1) / bs)); };
    auto gemm = [&](const bf16_t* A, const bf16_t* BwT, const float* bias,
                    const float* resid, int ldr, float* oF, int ldf,
                    bf16_t* oH, int ldh, int M, int Npad, int Nreal, int K,
                    int mode, int lda, int Cin, int IH, int IW, int OH, int OW,
                    int ksz, int pad, int relu) {
        dim3 g(Npad / 64, M / 64);
        k_conv_gemm<<<g, 256, 0, stream>>>(A, BwT, bias, resid, ldr, oF, ldf, oH, ldh,
                                           M, Npad, Nreal, K, mode, lda,
                                           Cin, IH, IW, OH, OW, ksz, pad, relu);
    };

    // ---- casts & weight transforms ----
    k_cast_bf16<<<blocks(33554432, 256), 256, 0, stream>>>(lve, lve_bf, 33554432);
    k_cast_bf16<<<blocks(98304, 256), 256, 0, stream>>>(text, txt_bf, 98304);
    k_cast_bf16<<<blocks(2097152, 256), 256, 0, stream>>>(qry, q2d_bf, 2097152);
    k_cast_bf16<<<blocks((size_t)512 * 256, 256), 256, 0, stream>>>(cb_top, cbH_top, (size_t)512 * 256);
    k_cast_bf16<<<blocks((size_t)512 * 512, 256), 256, 0, stream>>>(cb_bot, cbH_bot, (size_t)512 * 512);
    k_wt_conv <<<blocks((size_t)256 * 768, 256), 256, 0, stream>>>(btxt_w, wt_txt, 256, 768, 1, 1, 256);
    k_wt_conv <<<blocks((size_t)256 * 1024, 256), 256, 0, stream>>>(bvis_w, wt_vis, 256, 1024, 1, 1, 256);
    k_wt_conv <<<blocks((size_t)256 * 2304, 256), 256, 0, stream>>>(tb_w, wt_tb, 256, 256, 3, 3, 256);
    k_wt_conv <<<blocks((size_t)512 * 2304, 256), 256, 0, stream>>>(dtin_w, wt_dtin, 512, 256, 3, 3, 512);
    k_wt_conv <<<blocks((size_t)512 * 4608, 256), 256, 0, stream>>>(dtr1_w, wt_dtr1, 512, 512, 3, 3, 512);
    k_wt_conv <<<blocks((size_t)512 * 4608, 256), 256, 0, stream>>>(dtr2_w, wt_dtr2, 512, 512, 3, 3, 512);
    k_wt_convT<<<blocks((size_t)256 * 8192, 256), 256, 0, stream>>>(dtup_w, wt_dtup, 512, 256, 4, 4, 256);
    k_wt_convT<<<blocks((size_t)256 * 4096, 256), 256, 0, stream>>>(up_w, wt_up, 256, 256, 4, 4, 256);
    k_wt_conv <<<blocks((size_t)768 * 6912, 256), 256, 0, stream>>>(ds_w, wt_ds, 768, 768, 3, 3, 768);
    k_wt_conv <<<blocks((size_t)768 * 6912, 256), 256, 0, stream>>>(dbin_w, wt_dbin, 768, 768, 3, 3, 768);
    k_wt_conv <<<blocks((size_t)768 * 6912, 256), 256, 0, stream>>>(dbr1_w, wt_dbr1, 768, 768, 3, 3, 768);
    k_wt_conv <<<blocks((size_t)768 * 6912, 256), 256, 0, stream>>>(dbr2_w, wt_dbr2, 768, 768, 3, 3, 768);
    k_wt_convT<<<blocks((size_t)384 * 12288, 256), 256, 0, stream>>>(dbu1_w, wt_dbu1, 768, 384, 4, 4, 384);
    k_wt_convT<<<blocks((size_t)64 * 6144, 256), 256, 0, stream>>>(dbu2_w, wt_dbu2, 384, 3, 4, 4, 64);
    k_cnorm<<<blocks(512, 128), 128, 0, stream>>>(cb_top, 512, 256, cn_top);
    k_cnorm<<<blocks(512, 128), 128, 0, stream>>>(cb_bot, 512, 512, cn_bot);
    hipMemsetAsync(accb, 0, 1026 * 4, stream);

    // ---- bridges ----
    // txt = text @ bb_txt_w^T + b     [128,256]
    gemm(txt_bf, wt_txt, btxt_b, nullptr, 0, txtF, 256, nullptr, 0,
         128, 256, 256, 768, 0, 768, 0, 0, 0, 0, 0, 0, 0, 0);
    // z_e_bot pre-txt: 1x1 conv on visual grid -> fused left half (f32)
    gemm(lve_bf, wt_vis, bvis_b, nullptr, 0, fusedF, 512, nullptr, 0,
         32768, 256, 256, 1024, 0, 1024, 0, 0, 0, 0, 0, 0, 0, 0);
    k_txt_relu<<<blocks((size_t)32768 * 256, 256), 256, 0, stream>>>(fusedF, txtF, 256, 512, (size_t)32768 * 256);
    // z_e_top = conv3x3(q2d)          [8192,256]
    gemm(q2d_bf, wt_tb, tb_b, nullptr, 0, zetopF, 256, zetopH, 256,
         8192, 256, 256, 2304, 1, 0, 256, 8, 8, 8, 8, 3, 1, 0);

    // ---- quantize top ----
    gemm(zetopH, cbH_top, nullptr, nullptr, 0, scores, 512, nullptr, 0,
         8192, 512, 512, 256, 0, 256, 0, 0, 0, 0, 0, 0, 0, 0);
    k_argmin<<<blocks(8192 * 32, 256), 256, 0, stream>>>(scores, cn_top, 8192, 512, ids);
    k_count<<<blocks(8192, 256), 256, 0, stream>>>(ids, accb + 2, 8192);
    k_vq_gather<<<8192, 256, 0, stream>>>(ids, cb_top, 256, zetopF, 256,
                                          zqtopF, 256, zqtopH, 256, accb + 0);

    // ---- decoder top ----
    gemm(zqtopH, wt_dtin, dtin_b, nullptr, 0, hTop, 512, aTop, 512,
         8192, 512, 512, 2304, 1, 0, 256, 8, 8, 8, 8, 3, 1, 2);
    gemm(aTop, wt_dtr1, dtr1_b, nullptr, 0, nullptr, 0, bTop, 512,
         8192, 512, 512, 4608, 1, 0, 512, 8, 8, 8, 8, 3, 1, 1);
    gemm(bTop, wt_dtr2, dtr2_b, hTop, 512, nullptr, 0, cTop, 512,
         8192, 512, 512, 4608, 1, 0, 512, 8, 8, 8, 8, 3, 1, 2);
    // zq_t2b = convT(relu(h)) -> fused right half (f32)
    gemm(cTop, wt_dtup, dtup_b, nullptr, 0, fusedF + 256, 512, nullptr, 0,
         32768, 256, 256, 8192, 2, 0, 512, 8, 8, 16, 16, 4, 1, 0);
    k_cast_bf16<<<blocks((size_t)32768 * 512, 256), 256, 0, stream>>>(fusedF, fusedH, (size_t)32768 * 512);

    // ---- quantize bottom ----
    gemm(fusedH, cbH_bot, nullptr, nullptr, 0, scores, 512, nullptr, 0,
         32768, 512, 512, 512, 0, 512, 0, 0, 0, 0, 0, 0, 0, 0);
    k_argmin<<<blocks((size_t)32768 * 32, 256), 256, 0, stream>>>(scores, cn_bot, 32768, 512, ids);
    k_count<<<blocks(32768, 256), 256, 0, stream>>>(ids, accb + 514, 32768);
    k_vq_gather<<<32768, 256, 0, stream>>>(ids, cb_bot, 512, fusedF, 512,
                                           nullptr, 0, zqH + 256, 768, accb + 1);

    // ---- zq concat + downsize conv ----
    gemm(zqtopH, wt_up, up_b, nullptr, 0, nullptr, 0, zqH, 768,
         32768, 256, 256, 4096, 2, 0, 256, 8, 8, 16, 16, 4, 1, 0);
    gemm(zqH, wt_ds, ds_b, nullptr, 0, nullptr, 0, zq2H, 768,
         25088, 768, 768, 6912, 1, 0, 768, 16, 16, 14, 14, 3, 0, 0);

    // ---- decoder bottom ----
    gemm(zq2H, wt_dbin, dbin_b, nullptr, 0, hBot, 768, aBot, 768,
         25088, 768, 768, 6912, 1, 0, 768, 14, 14, 14, 14, 3, 1, 2);
    gemm(aBot, wt_dbr1, dbr1_b, nullptr, 0, nullptr, 0, bBot, 768,
         25088, 768, 768, 6912, 1, 0, 768, 14, 14, 14, 14, 3, 1, 1);
    gemm(bBot, wt_dbr2, dbr2_b, hBot, 768, nullptr, 0, cBot, 768,
         25088, 768, 768, 6912, 1, 0, 768, 14, 14, 14, 14, 3, 1, 2);
    gemm(cBot, wt_dbu1, dbu1_b, nullptr, 0, nullptr, 0, up1H, 384,
         100352, 384, 384, 12288, 2, 0, 768, 14, 14, 28, 28, 4, 1, 1);
    gemm(up1H, wt_dbu2, dbu2_b, nullptr, 0, reconF, 3, nullptr, 0,
         401408, 64, 3, 6144, 2, 0, 384, 28, 28, 56, 56, 4, 1, 0);

    // ---- outputs ----
    k_nchw<<<blocks(401408 * 3, 256), 256, 0, stream>>>(reconF, out, 401408 * 3);
    k_finalize<<<1, 512, 0, stream>>>(accb, out + 1204224);
}
